// DRCN_58695023067404
// MI455X (gfx1250) — matrix operand order repaired and checked by reading, not compiled, not run on hardware
//
#include <hip/hip_runtime.h>
#include <math.h>

// Problem dims
constexpr int kB = 256, kL = 256, kE = 300, kH = 100, kG = 400; // kG = 4H

typedef __attribute__((ext_vector_type(16))) _Float16 v16h;
typedef __attribute__((ext_vector_type(2)))  _Float16 h2;
typedef __attribute__((ext_vector_type(8)))  float    v8f;
typedef __attribute__((ext_vector_type(2)))  float    f2;

__device__ __forceinline__ int imin(int a, int b) { return a < b ? a : b; }

// ---------------- WMMA fragment K mappings (wave32, 16x16x32 f16) ----------
// A (16xK): lane = half*16 + m ; halves hold K = e + 16*half
__device__ __forceinline__ int a_kmap(int e, int half) {
  return e + (half << 4);
}
// B (Kx16): lane = half*16 + n ; halves hold K = e + 16*half
__device__ __forceinline__ int b_kmap(int e, int half) {
  return e + (half << 4);
}
__device__ __forceinline__ float sigf(float x) { return 1.f / (1.f + __expf(-x)); }

__device__ __forceinline__ float wave_max(float v) {
#pragma unroll
  for (int o = 16; o > 0; o >>= 1) v = fmaxf(v, __shfl_xor(v, o, 32));
  return v;
}
__device__ __forceinline__ float wave_sum(float v) {
#pragma unroll
  for (int o = 16; o > 0; o >>= 1) v += __shfl_xor(v, o, 32);
  return v;
}

// ---------------- Generic batched WMMA GEMM --------------------------------
// C[b] = act( A[b] (MxK) @ B[b] (KxN) + bias ), f32 in/out, f16 compute.
// 8 waves/block, 4x2 -> 64x32 C macro-tile; K staged in chunks of 64
// (2 WMMAs per barrier pair). act: 0 none, 1 tanh, 2 relu.
constexpr int KC = 64;          // K chunk
constexpr int LDK = KC + 2;     // padded LDS K stride (even, keeps h2 aligned)

__global__ __launch_bounds__(256) void gemm_wmma_kernel(
    const float* __restrict__ A, int lda, long long strideA,
    const float* __restrict__ Bm, int ldb, long long strideB, int transB,
    float* __restrict__ C, int ldc, long long strideC,
    const float* __restrict__ bias, int M, int N, int K, int act)
{
  __shared__ _Float16 shA[64 * LDK];   // A  tile [m][k]
  __shared__ _Float16 shBT[32 * LDK];  // B^T tile [n][k]

  const int bz = blockIdx.z;
  A  += (long long)bz * strideA;
  Bm += (long long)bz * strideB;
  C  += (long long)bz * strideC;

  const int m0 = blockIdx.y * 64;
  const int n0 = blockIdx.x * 32;
  const int tid  = threadIdx.x;
  const int wave = tid >> 5;
  const int lane = tid & 31;
  const int mt = wave >> 1;         // 0..3
  const int nt = wave & 1;          // 0..1
  const int half = lane >> 4;
  const int l16  = lane & 15;
  const int mMax = M - 1, nMax = N - 1;
  const int kEven = (K - 1) & ~1;   // even clamp for contiguous-k b64 loads
  const int nEven = (N - 1) & ~1;   // even clamp for contiguous-n b64 loads

  v8f acc = {0.f,0.f,0.f,0.f,0.f,0.f,0.f,0.f};

  for (int k0 = 0; k0 < K; k0 += KC) {
    // ---- stage A (64 x KC): phase 1 = unconditional clamped b64 loads
    f2 va[8];
#pragma unroll
    for (int u = 0; u < 8; ++u) {
      int pi = tid + u * 256;                 // pair index (2048 pairs)
      int r = pi >> 5, c = (pi & 31) * 2;     // k fast -> coalesced
      int gm = m0 + r, gk = k0 + c;
      va[u] = *(const f2*)(A + (long long)imin(gm, mMax) * lda + imin(gk, kEven));
    }
    // ---- stage B^T (32 x KC)
    f2 vb[4];
    if (transB) {
#pragma unroll
      for (int u = 0; u < 4; ++u) {
        int pi = tid + u * 256;               // 1024 pairs, k fast
        int nn = pi >> 5, c = (pi & 31) * 2;
        int gk = k0 + c, gn = n0 + nn;
        vb[u] = *(const f2*)(Bm + (long long)imin(gn, nMax) * ldb + imin(gk, kEven));
      }
    } else {
#pragma unroll
      for (int u = 0; u < 4; ++u) {
        int pi = tid + u * 256;               // n fast (contiguous in B src)
        int nn = (pi & 15) * 2, c = pi >> 4;
        int gk = k0 + c, gn = n0 + nn;
        vb[u] = *(const f2*)(Bm + (long long)imin(gk, K - 1) * ldb + imin(gn, nEven));
      }
    }
    // ---- phase 2: mask (x * {0,1}, not foldable) + paired f16 LDS stores
#pragma unroll
    for (int u = 0; u < 8; ++u) {
      int pi = tid + u * 256;
      int r = pi >> 5, c = (pi & 31) * 2;
      int gm = m0 + r, gk = k0 + c;
      float mrow = (gm < M) ? 1.f : 0.f;
      float mk0 = (gk < K) ? mrow : 0.f;
      float mk1 = (gk + 1 < K) ? mrow : 0.f;
      h2 hv; hv[0] = (_Float16)(va[u][0] * mk0); hv[1] = (_Float16)(va[u][1] * mk1);
      *(h2*)&shA[r * LDK + c] = hv;
    }
#pragma unroll
    for (int u = 0; u < 4; ++u) {
      int pi = tid + u * 256;
      h2 hv;
      if (transB) {
        int nn = pi >> 5, c = (pi & 31) * 2;
        int gk = k0 + c, gn = n0 + nn;
        float mrow = (gn < N) ? 1.f : 0.f;
        hv[0] = (_Float16)(vb[u][0] * ((gk < K) ? mrow : 0.f));
        hv[1] = (_Float16)(vb[u][1] * ((gk + 1 < K) ? mrow : 0.f));
        *(h2*)&shBT[nn * LDK + c] = hv;
      } else {
        int nn = (pi & 15) * 2, c = pi >> 4;
        int gk = k0 + c, gn = n0 + nn;
        float mrow = (gk < K) ? 1.f : 0.f;
        hv[0] = (_Float16)(vb[u][0] * ((gn < N) ? mrow : 0.f));
        hv[1] = (_Float16)(vb[u][1] * ((gn + 1 < N) ? mrow : 0.f));
        // transposed scatter: two half stores (different n rows)
        shBT[nn * LDK + c] = hv[0];
        shBT[(nn + 1) * LDK + c] = hv[1];
      }
    }
    __syncthreads();

    if (k0 + KC < K) {  // hint next A tile toward the caches
      int pr = m0 + (tid >> 2);
      if (pr < M) __builtin_prefetch(&A[(long long)pr * lda + k0 + KC], 0, 1);
    }

    const h2* pA = (const h2*)&shA[(mt * 16 + l16) * LDK];
    const h2* pB = (const h2*)&shBT[(nt * 16 + l16) * LDK];
#pragma unroll
    for (int ks = 0; ks < 2; ++ks) {
      v16h af, bf;
#pragma unroll
      for (int v = 0; v < 8; ++v) {
        h2 a2 = pA[(ks * 32 + a_kmap(2 * v, half)) >> 1];
        h2 b2 = pB[(ks * 32 + b_kmap(2 * v, half)) >> 1];
        af[2 * v] = a2[0]; af[2 * v + 1] = a2[1];
        bf[2 * v] = b2[0]; bf[2 * v + 1] = b2[1];
      }
      acc = __builtin_amdgcn_wmma_f32_16x16x32_f16(false, af, false, bf,
                                                   (short)0, acc, false, false);
    }
    __syncthreads();
  }

  const int gn = n0 + nt * 16 + l16;
#pragma unroll
  for (int r = 0; r < 8; ++r) {
    int gm = m0 + mt * 16 + r + half * 8;
    if (gm < M && gn < N) {
      float v = acc[r];
      if (bias) v += bias[gn];
      if (act == 1) v = tanhf(v);
      else if (act == 2) v = fmaxf(v, 0.f);
      C[(long long)gm * ldc + gn] = v;
    }
  }
}

// ---------------- Persistent WMMA LSTM -------------------------------------
// Block owns 16 batch rows. In LDS for all 2*L steps: Whh^T (f16 [400][128]),
// h (f16 [16][128]), gates (f32 [16][400]), c (f32 [16][100]), and the
// time-constant input projection pre (f32 [16][400], staged per sequence).
// Sequence q continues from p's final state (faithful to the reference).
constexpr int LSTM_SMEM = 400 * 128 * 2   // Whh^T f16
                        + 16 * 128 * 2    // h     f16
                        + 16 * 400 * 4    // g     f32
                        + 16 * 100 * 4    // c     f32
                        + 16 * 400 * 4;   // pre   f32

__global__ __launch_bounds__(256) void lstm_wmma_kernel(
    const float* __restrict__ pre_p, const float* __restrict__ pre_q,
    const float* __restrict__ Whh,
    float* __restrict__ hsP, int ldhP,
    float* __restrict__ hsQ, int ldhQ)
{
  extern __shared__ char smem[];
  _Float16* sWT = (_Float16*)smem;                    // [400][128] (n-major)
  _Float16* sH  = (_Float16*)(smem + 102400);         // [16][128]
  float* sG = (float*)(smem + 102400 + 4096);                 // [16][400]
  float* sC = (float*)(smem + 102400 + 4096 + 25600);         // [16][100]
  float* sP = (float*)(smem + 102400 + 4096 + 25600 + 6400);  // [16][400]

  const int tid = threadIdx.x;
  const int b0  = blockIdx.x * 16;

  // Whh [100][400] -> sWT [n][kpad=128]; unconditional coalesced loads,
  // pad rows zero-filled separately (no conditional loads anywhere).
  for (int i = tid; i < 100 * 400; i += 256) {   // i = k*400 + n
    int k = i / 400, n = i % 400;
    sWT[n * 128 + k] = (_Float16)Whh[i];
  }
  for (int i = tid; i < 28 * 400; i += 256) {    // pad k = 100..127
    int k = 100 + i / 400, n = i % 400;
    sWT[n * 128 + k] = (_Float16)0.f;
  }
  for (int i = tid; i < 16 * 128; i += 256)
    sH[i] = (_Float16)(((i & 127) < kH) ? 1.f : 0.f);   // h0 = 1
  for (int i = tid; i < 16 * 100; i += 256) sC[i] = 1.f; // c0 = 1

  const int wave = tid >> 5, lane = tid & 31;
  const int half = lane >> 4, l16 = lane & 15;

  for (int seq = 0; seq < 2; ++seq) {
    const float* pre = seq ? pre_q : pre_p;
    float* hs = seq ? hsQ : hsP;
    const int ldh = seq ? ldhQ : ldhP;

    // stage pre for our 16 rows (contiguous, coalesced)
    for (int i = tid; i < 16 * kG; i += 256) sP[i] = pre[(long long)b0 * kG + i];
    __syncthreads();

    for (int t = 0; t < kL; ++t) {
      // g = pre + h @ Whh via WMMA; 25 N-tiles of 16, K = 128 (4 steps)
      for (int ntile = wave; ntile < 25; ntile += 8) {
        const int nc0 = ntile * 16;
        v8f acc;
#pragma unroll
        for (int r = 0; r < 8; ++r)
          acc[r] = sP[(r + half * 8) * kG + nc0 + l16];
        const h2* pA = (const h2*)&sH[l16 * 128];
        const h2* pB = (const h2*)&sWT[(nc0 + l16) * 128];
#pragma unroll
        for (int ks = 0; ks < 4; ++ks) {
          v16h af, bf;
#pragma unroll
          for (int v = 0; v < 8; ++v) {
            h2 a2 = pA[(ks * 32 + a_kmap(2 * v, half)) >> 1];
            h2 b2 = pB[(ks * 32 + b_kmap(2 * v, half)) >> 1];
            af[2 * v] = a2[0]; af[2 * v + 1] = a2[1];
            bf[2 * v] = b2[0]; bf[2 * v + 1] = b2[1];
          }
          acc = __builtin_amdgcn_wmma_f32_16x16x32_f16(false, af, false, bf,
                                                       (short)0, acc, false, false);
        }
#pragma unroll
        for (int r = 0; r < 8; ++r)
          sG[(r + half * 8) * kG + nc0 + l16] = acc[r];
      }
      __syncthreads();

      // gate fusion: i,f,g,o order (torch), update c,h; emit h
      for (int i = tid; i < 16 * kH; i += 256) {
        int row = i / kH, j = i % kH;
        float gi = sG[row * kG + j];
        float gf = sG[row * kG + 100 + j];
        float gg = sG[row * kG + 200 + j];
        float go = sG[row * kG + 300 + j];
        float c = sigf(gf) * sC[row * kH + j] + sigf(gi) * tanhf(gg);
        float h = sigf(go) * tanhf(c);
        sC[row * kH + j] = c;
        sH[row * 128 + j] = (_Float16)h;
        hs[((long long)(b0 + row) * kL + t) * ldh + j] = h;
      }
      __syncthreads();
    }
  }
}

// ---------------- Utility kernels ------------------------------------------
__global__ void gather_embed_kernel(const int* __restrict__ idx,
                                    const float* __restrict__ embed,
                                    float* __restrict__ out)
{
  long long total = (long long)kB * kL * kE;
  for (long long t = (long long)blockIdx.x * blockDim.x + threadIdx.x; t < total;
       t += (long long)gridDim.x * blockDim.x) {
    int e = (int)(t % kE);
    long long r = t / kE;
    out[t] = embed[(long long)idx[r] * kE + e];
  }
}

__global__ void vecadd_kernel(const float* a, const float* b, float* c, int n) {
  int i = blockIdx.x * blockDim.x + threadIdx.x;
  if (i < n) c[i] = a[i] + b[i];
}

__global__ void rownorm_kernel(const float* __restrict__ x, int ld, int cols,
                               float* __restrict__ out, int rows)
{
  for (int r = blockIdx.x * blockDim.x + threadIdx.x; r < rows;
       r += gridDim.x * blockDim.x) {
    const float* p = x + (long long)r * ld;
    float s = 0.f;
    for (int c = 0; c < cols; ++c) { float v = p[c]; s += v * v; }
    out[r] = sqrtf(s);
  }
}

__global__ void attnorm_kernel(float* __restrict__ att,
                               const float* __restrict__ pn,
                               const float* __restrict__ qn)
{
  long long total = (long long)kB * kL * kL;
  for (long long t = (long long)blockIdx.x * blockDim.x + threadIdx.x; t < total;
       t += (long long)gridDim.x * blockDim.x) {
    int j = (int)(t % kL);
    long long r = t / kL;
    int i = (int)(r % kL);
    int b = (int)(r / kL);
    float d = fmaxf(pn[b * kL + i] * qn[b * kL + j], 1e-8f);
    att[t] /= d;
  }
}

__global__ __launch_bounds__(256) void softmax_rows_kernel(
    const float* __restrict__ att, float* __restrict__ out, int nrows)
{
  int wave = threadIdx.x >> 5, lane = threadIdx.x & 31;
  int row = blockIdx.x * 8 + wave;
  if (row >= nrows) return;
  const float* a = att + (long long)row * kL;
  float m = -1e30f;
  for (int j = lane; j < kL; j += 32) m = fmaxf(m, a[j]);
  m = wave_max(m);
  float e[8]; float s = 0.f;
  for (int j = lane, t = 0; j < kL; j += 32, ++t) { e[t] = __expf(a[j] - m); s += e[t]; }
  s = wave_sum(s);
  float inv = 1.f / s;
  float* o = out + (long long)row * kL;
  for (int j = lane, t = 0; j < kL; j += 32, ++t) o[j] = e[t] * inv;
}

__global__ __launch_bounds__(256) void softmax_cols_kernel(
    const float* __restrict__ att, float* __restrict__ out, int ncols)
{
  int wave = threadIdx.x >> 5, lane = threadIdx.x & 31;
  int col = blockIdx.x * 8 + wave;
  if (col >= ncols) return;
  int b = col / kL, j = col % kL;
  const float* a = att + (long long)b * kL * kL + j;
  float m = -1e30f;
  for (int i = lane; i < kL; i += 32) m = fmaxf(m, a[(long long)i * kL]);
  m = wave_max(m);
  float e[8]; float s = 0.f;
  for (int i = lane, t = 0; i < kL; i += 32, ++t) { e[t] = __expf(a[(long long)i * kL] - m); s += e[t]; }
  s = wave_sum(s);
  float inv = 1.f / s;
  float* o = out + (long long)b * kL * kL + j;
  for (int i = lane, t = 0; i < kL; i += 32, ++t) o[(long long)i * kL] = e[t] * inv;
}

__global__ void copych_kernel(const float* __restrict__ src, int lds_,
                              float* __restrict__ dst, int ldd,
                              long long rows, int cols)
{
  long long total = rows * cols;
  for (long long t = (long long)blockIdx.x * blockDim.x + threadIdx.x; t < total;
       t += (long long)gridDim.x * blockDim.x) {
    int c = (int)(t % cols);
    long long r = t / cols;
    dst[r * ldd + c] = src[r * lds_ + c];
  }
}

__global__ void maxpool_kernel(const float* __restrict__ x, int C,
                               float* __restrict__ out)
{
  int total = kB * C;
  for (int t = blockIdx.x * blockDim.x + threadIdx.x; t < total;
       t += gridDim.x * blockDim.x) {
    int c = t % C, b = t / C;
    const float* p = x + (long long)b * kL * C + c;
    float m = -1e30f;
    for (int l = 0; l < kL; ++l) m = fmaxf(m, p[(long long)l * C]);
    out[t] = m;
  }
}

__global__ void buildvec_kernel(const float* __restrict__ vp,
                                const float* __restrict__ vq,
                                float* __restrict__ vec)
{
  const int C = 13 * kH;
  int total = kB * 5 * C;
  for (int t = blockIdx.x * blockDim.x + threadIdx.x; t < total;
       t += gridDim.x * blockDim.x) {
    int c = t % (5 * C), b = t / (5 * C);
    int seg = c / C, cc = c % C;
    float a = vp[b * C + cc], q = vq[b * C + cc], v;
    switch (seg) {
      case 0: v = a; break;
      case 1: v = q; break;
      case 2: v = a + q; break;
      case 3: v = a - q; break;
      default: v = fabsf(a - q); break;
    }
    vec[(long long)b * 5 * C + c] = v;
  }
}

__global__ void softmax2_kernel(const float* __restrict__ logits,
                                float* __restrict__ out)
{
  int b = blockIdx.x * blockDim.x + threadIdx.x;
  if (b >= kB) return;
  float a = logits[b * 2], c = logits[b * 2 + 1];
  float m = fmaxf(a, c);
  float ea = __expf(a - m), ec = __expf(c - m);
  float inv = 1.f / (ea + ec);
  out[b * 2] = ea * inv;
  out[b * 2 + 1] = ec * inv;
}

// ---------------- Host orchestration ---------------------------------------
static void launch_gemm(hipStream_t s,
                        const float* A, int lda, long long sA,
                        const float* Bm, int ldb, long long sB, int transB,
                        float* C, int ldc, long long sC, const float* bias,
                        int M, int N, int K, int act, int batch)
{
  dim3 grid((N + 31) / 32, (M + 63) / 64, batch);
  gemm_wmma_kernel<<<grid, dim3(256), 0, s>>>(A, lda, sA, Bm, ldb, sB, transB,
                                              C, ldc, sC, bias, M, N, K, act);
}

static void run_layer(hipStream_t s,
                      const float* inP, const float* inQ, int Cin,
                      float* outP, float* outQ, int Cout,
                      const float* Wih, const float* Whh,
                      const float* bih, const float* bhh,
                      float* pre_p, float* pre_q, float* pn, float* qn,
                      float* att, float* sm, float* bsum)
{
  const long long BL = (long long)kB * kL;
  vecadd_kernel<<<2, 256, 0, s>>>(bih, bhh, bsum, kG);
  // pre = x[:,0,:] @ Wih + (bih + bhh)  -- constant over time (faithful quirk)
  launch_gemm(s, inP, kL * Cin, 0, Wih, kG, 0, 0, pre_p, kG, 0, bsum, kB, kG, Cin, 0, 1);
  launch_gemm(s, inQ, kL * Cin, 0, Wih, kG, 0, 0, pre_q, kG, 0, bsum, kB, kG, Cin, 0, 1);

  lstm_wmma_kernel<<<kB / 16, 256, LSTM_SMEM, s>>>(pre_p, pre_q, Whh,
                                                   outP, Cout, outQ, Cout);

  rownorm_kernel<<<256, 256, 0, s>>>(outP, Cout, kH, pn, kB * kL);
  rownorm_kernel<<<256, 256, 0, s>>>(outQ, Cout, kH, qn, kB * kL);

  // dots[b] = hs_p[b] @ hs_q[b]^T  (batched, transB)
  launch_gemm(s, outP, Cout, (long long)kL * Cout,
              outQ, Cout, (long long)kL * Cout, 1,
              att, kL, (long long)kL * kL, nullptr, kL, kL, kH, 0, kB);
  attnorm_kernel<<<4096, 256, 0, s>>>(att, pn, qn);

  softmax_rows_kernel<<<(kB * kL) / 8, 256, 0, s>>>(att, sm, kB * kL);
  // a_p written straight into concat slice (channel offset H)
  launch_gemm(s, sm, kL, (long long)kL * kL,
              outQ, Cout, (long long)kL * Cout, 0,
              outP + kH, Cout, (long long)kL * Cout, nullptr, kL, kH, kL, 0, kB);

  softmax_cols_kernel<<<(kB * kL) / 8, 256, 0, s>>>(att, sm, kB * kL);
  launch_gemm(s, sm, kL, (long long)kL * kL,
              outP, Cout, (long long)kL * Cout, 0,
              outQ + kH, Cout, (long long)kL * Cout, nullptr, kL, kH, kL, 0, kB);

  // concat previous input at channel offset 2H
  copych_kernel<<<4096, 256, 0, s>>>(inP, Cin, outP + 2 * kH, Cout, BL, Cin);
  copych_kernel<<<4096, 256, 0, s>>>(inQ, Cin, outQ + 2 * kH, Cout, BL, Cin);
}

static void run_ae(hipStream_t s, const float* in, int Cin,
                   const float* w1, const float* b1,
                   const float* w2, const float* b2,
                   float* t1, float* out, int Cout)
{
  launch_gemm(s, in, Cin, 0, w1, 200, 0, 0, t1, 200, 0, b1,
              kB * kL, 200, Cin, 1, 1);
  launch_gemm(s, t1, 200, 0, w2, Cout, 0, 0, out, Cout, 0, b2,
              kB * kL, Cout, 200, 1, 1);
}

extern "C" void kernel_launch(void* const* d_in, const int* in_sizes, int n_in,
                              void* d_out, int out_size, void* d_ws, size_t ws_size,
                              hipStream_t stream)
{
  (void)in_sizes; (void)n_in; (void)out_size;
  const int* p = (const int*)d_in[0];
  const int* q = (const int*)d_in[1];
  const float* embed = (const float*)d_in[2];
  const float *Wih[5], *Whh[5], *bih[5], *bhh[5];
  for (int i = 0; i < 5; ++i) {
    Wih[i] = (const float*)d_in[3 + 4 * i];
    Whh[i] = (const float*)d_in[4 + 4 * i];
    bih[i] = (const float*)d_in[5 + 4 * i];
    bhh[i] = (const float*)d_in[6 + 4 * i];
  }
  const float* a1w1 = (const float*)d_in[23]; const float* a1b1 = (const float*)d_in[24];
  const float* a1w2 = (const float*)d_in[25]; const float* a1b2 = (const float*)d_in[26];
  const float* a2w1 = (const float*)d_in[27]; const float* a2b1 = (const float*)d_in[28];
  const float* a2w2 = (const float*)d_in[29]; const float* a2b2 = (const float*)d_in[30];
  const float* dw1  = (const float*)d_in[31]; const float* db1  = (const float*)d_in[32];
  const float* dw2  = (const float*)d_in[33]; const float* db2  = (const float*)d_in[34];
  const float* dw3  = (const float*)d_in[35]; const float* db3  = (const float*)d_in[36];

  float* ws = (float*)d_ws;
  size_t off = 0;
  auto alloc = [&](size_t n) { float* r = ws + off; off += n; return r; };
  const size_t BL = (size_t)kB * kL;

  float* xAp = alloc(BL * 1300); float* xAq = alloc(BL * 1300);
  float* xBp = alloc(BL * 1300); float* xBq = alloc(BL * 1300);
  float* att = alloc(BL * kL);
  float* sm  = alloc(BL * kL);
  float* t1  = alloc(BL * 200);
  float* pre_p = alloc((size_t)kB * kG); float* pre_q = alloc((size_t)kB * kG);
  float* pn = alloc(BL); float* qn = alloc(BL);
  float* bsum = alloc(kG);
  float* vp = alloc((size_t)kB * 1300); float* vq = alloc((size_t)kB * 1300);
  float* vec = alloc((size_t)kB * 6500);
  float* h1 = alloc((size_t)kB * 1000); float* h2 = alloc((size_t)kB * 1000);
  float* logits = alloc((size_t)kB * 2);
  if (off * sizeof(float) > ws_size) return;  // deterministic no-op if scratch short

  // embeddings, layout [B, L, C] throughout
  gather_embed_kernel<<<2048, 256, 0, stream>>>(p, embed, xAp);
  gather_embed_kernel<<<2048, 256, 0, stream>>>(q, embed, xAq);

  run_layer(stream, xAp, xAq, 300, xBp, xBq, 500, Wih[0], Whh[0], bih[0], bhh[0],
            pre_p, pre_q, pn, qn, att, sm, bsum);
  run_layer(stream, xBp, xBq, 500, xAp, xAq, 700, Wih[1], Whh[1], bih[1], bhh[1],
            pre_p, pre_q, pn, qn, att, sm, bsum);
  run_layer(stream, xAp, xAq, 700, xBp, xBq, 900, Wih[2], Whh[2], bih[2], bhh[2],
            pre_p, pre_q, pn, qn, att, sm, bsum);

  run_ae(stream, xBp, 900, a1w1, a1b1, a1w2, a1b2, t1, xAp, 900);
  run_ae(stream, xBq, 900, a1w1, a1b1, a1w2, a1b2, t1, xAq, 900);

  run_layer(stream, xAp, xAq, 900, xBp, xBq, 1100, Wih[3], Whh[3], bih[3], bhh[3],
            pre_p, pre_q, pn, qn, att, sm, bsum);
  run_layer(stream, xBp, xBq, 1100, xAp, xAq, 1300, Wih[4], Whh[4], bih[4], bhh[4],
            pre_p, pre_q, pn, qn, att, sm, bsum);

  run_ae(stream, xAp, 1300, a2w1, a2b1, a2w2, a2b2, t1, xBp, 1300);
  run_ae(stream, xAq, 1300, a2w1, a2b1, a2w2, a2b2, t1, xBq, 1300);

  maxpool_kernel<<<1024, 256, 0, stream>>>(xBp, 1300, vp);
  maxpool_kernel<<<1024, 256, 0, stream>>>(xBq, 1300, vq);
  buildvec_kernel<<<2048, 256, 0, stream>>>(vp, vq, vec);

  launch_gemm(stream, vec, 6500, 0, dw1, 1000, 0, 0, h1, 1000, 0, db1,
              kB, 1000, 6500, 2, 1);
  launch_gemm(stream, h1, 1000, 0, dw2, 1000, 0, 0, h2, 1000, 0, db2,
              kB, 1000, 1000, 2, 1);
  launch_gemm(stream, h2, 1000, 0, dw3, 2, 0, 0, logits, 2, 0, db3,
              kB, 2, 1000, 0, 1);
  softmax2_kernel<<<1, 256, 0, stream>>>(logits, (float*)d_out);
}
